// WeightOnlyInt4Linear_18811956757004
// MI455X (gfx1250) — matrix operand order repaired and checked by reading, not compiled, not run on hardware
//
#include <hip/hip_runtime.h>
#include <hip/hip_bf16.h>
#include <stdint.h>

#define TOKENS 8192
#define IN_F   4096
#define OUT_F  11008

typedef __attribute__((ext_vector_type(16))) __bf16 v16bf;
typedef __attribute__((ext_vector_type(8)))  __bf16 bf16x8;
typedef __attribute__((ext_vector_type(8)))  float  v8f;

// CDNA5 async global->LDS DMA (GV mode, 16B per lane), ASYNCcnt-tracked.
// ldsoff = low 32 bits of generic pointer to __shared__ (LDS aperture offset).
__device__ __forceinline__ void async_copy_b128(uint32_t ldsoff, const void* gaddr) {
    asm volatile("global_load_async_to_lds_b128 %0, %1, off"
                 :: "v"(ldsoff), "v"(gaddr) : "memory");
}

// =====================================================================
// Pass 1: dequantize int4 codes -> bf16, stored K-major: wP[k][o].
// Each weight dequantized exactly ONCE (fused design would redo it 64x
// and be VALU-bound at ~20x the HBM floor).
// =====================================================================
#define P1_LD (128 + 8)

__global__ __launch_bounds__(256)
void dequant_transpose(const int*    __restrict__ qw,
                       const __bf16* __restrict__ sz,
                       __bf16*       __restrict__ wP)
{
    __shared__ __bf16 Ls[128 * P1_LD];     // [k][o]
    const int t  = threadIdx.x;
    const int o0 = blockIdx.x * 128;
    const int k0 = blockIdx.y * 128;
    const __bf16* szg = sz + (size_t)blockIdx.y * OUT_F * 2;  // group == k-tile

#pragma unroll 4
    for (int i = 0; i < 64; ++i) {
        int e = t + i * 256;               // 0..16383
        int o = e >> 7;                    // uniform per wave-instruction
        int k = e & 127;                   // lanes consecutive in k -> coalesced
        int q = qw[(size_t)(o0 + o) * IN_F + k0 + k];
        uint32_t p = *(const uint32_t*)(szg + (size_t)(o0 + o) * 2);
        union { uint32_t u; float f; } us, uz;
        us.u = p << 16;                    // scale (low bf16)
        uz.u = p & 0xffff0000u;            // zero  (high bf16)
        Ls[k * P1_LD + o] = (__bf16)((float)(q - 8) * us.f + uz.f);
    }
    __syncthreads();

#pragma unroll
    for (int i = 0; i < 8; ++i) {
        int c  = t + i * 256;              // 0..2047 b128 chunks
        int k  = c >> 4;
        int o8 = (c & 15) << 3;
        bf16x8 v = *(const bf16x8*)(Ls + k * P1_LD + o8);
        *(bf16x8*)(wP + (size_t)(k0 + k) * OUT_F + o0 + o8) = v;
    }
}

// =====================================================================
// Pass 2: bf16 GEMM out = x * wP^T (wP K-major), 256x256x64 tiles,
// 512 threads = 16 waves (wave tile 64x64 = 4x4 WMMA), double-buffered
// LDS filled by GLOBAL_LOAD_ASYNC_TO_LDS_B128 overlapping the WMMAs.
// =====================================================================
#define BM2 256
#define BN2 256
#define BK2 64
#define LDA2 (BK2 + 8)    // 72  bf16 pitch
#define LDB2 (BN2 + 8)    // 264 bf16 pitch: frag bank stride 132 DW == 4 mod 64
#define NT2  (IN_F / BK2) // 64 K-steps

__global__ __launch_bounds__(512)
void gemm_bf16_wmma_async(const __bf16* __restrict__ x,
                          const __bf16* __restrict__ wP,
                          __bf16*       __restrict__ out)
{
    __shared__ __bf16 Xs[2][BM2 * LDA2];   // [m][k], double buffered
    __shared__ __bf16 Ws[2][BK2 * LDB2];   // [k][n], double buffered

    const int t    = threadIdx.x;
    const int lane = t & 31;
    const int wid  = t >> 5;            // 0..15
    const int wrow = wid >> 2;          // 0..3 -> M offset *64
    const int wcol = wid & 3;           // 0..3 -> N offset *64

    const int bm0 = blockIdx.x * BM2;   // M fast -> W strip stays hot in L2
    const int bn0 = blockIdx.y * BN2;

    // per-thread copy coordinates (fixed across K-steps)
    const int xrow  = (t + 0 * 512) >> 3;        // recomputed per i below
    (void)xrow;
    const uint32_t xs_base = (uint32_t)(uintptr_t)(const void*)&Xs[0][0];
    const uint32_t ws_base = (uint32_t)(uintptr_t)(const void*)&Ws[0][0];
    const uint32_t xs_bufstride = (uint32_t)(BM2 * LDA2 * sizeof(__bf16));
    const uint32_t ws_bufstride = (uint32_t)(BK2 * LDB2 * sizeof(__bf16));

    // issue one tile's async copies: 4 X chunks + 4 W chunks per thread
    auto issue_tile = [&](int kb, int buf) {
        const int k0 = kb * BK2;
#pragma unroll
        for (int i = 0; i < 4; ++i) {
            int c    = t + i * 512;            // 0..2047
            int row  = c >> 3;                 // 0..255
            int col8 = (c & 7) << 3;           // 0..56
            async_copy_b128(xs_base + buf * xs_bufstride
                                + (uint32_t)((row * LDA2 + col8) * sizeof(__bf16)),
                            x + (size_t)(bm0 + row) * IN_F + k0 + col8);
        }
#pragma unroll
        for (int i = 0; i < 4; ++i) {
            int c  = t + i * 512;
            int k  = c >> 5;                   // 0..63
            int n8 = (c & 31) << 3;            // 0..248
            async_copy_b128(ws_base + buf * ws_bufstride
                                + (uint32_t)((k * LDB2 + n8) * sizeof(__bf16)),
                            wP + (size_t)(k0 + k) * OUT_F + bn0 + n8);
        }
    };

    v8f acc[4][4];
#pragma unroll
    for (int mi = 0; mi < 4; ++mi)
#pragma unroll
        for (int ni = 0; ni < 4; ++ni) {
            v8f z = {0.f, 0.f, 0.f, 0.f, 0.f, 0.f, 0.f, 0.f};
            acc[mi][ni] = z;
        }

    issue_tile(0, 0);

    for (int kb = 0; kb < NT2; ++kb) {
        const int buf = kb & 1;

        if (kb + 1 < NT2) {
            issue_tile(kb + 1, buf ^ 1);
            // 8 newest ops (next tile) may remain; older 8 (this tile) done.
            asm volatile("s_wait_asynccnt 0x8" ::: "memory");
        } else {
            asm volatile("s_wait_asynccnt 0x0" ::: "memory");
        }
        __syncthreads();   // tile kb resident in LDS for all waves

#pragma unroll
        for (int kk = 0; kk < BK2 / 32; ++kk) {
            const int ahalf = (lane >> 4) << 3;   // A: lane<16 K{0..7,16..23}; else {8..15,24..31}
            v16bf a[4], b[4];
#pragma unroll
            for (int mi = 0; mi < 4; ++mi) {
                int row = wrow * 64 + mi * 16 + (lane & 15);
                const __bf16* p = &Xs[buf][row * LDA2 + kk * 32 + ahalf];
                bf16x8 lo = *(const bf16x8*)(p);
                bf16x8 hi = *(const bf16x8*)(p + 16);
                a[mi] = __builtin_shufflevector(lo, hi,
                        0,1,2,3,4,5,6,7,8,9,10,11,12,13,14,15);
            }
#pragma unroll
            for (int ni = 0; ni < 4; ++ni) {
                int krow = kk * 32 + ahalf;       // B: lane = N column, K same order as A
                const __bf16* p = &Ws[buf][krow * LDB2 + wcol * 64 + ni * 16 + (lane & 15)];
#pragma unroll
                for (int s = 0; s < 16; ++s)
                    b[ni][s] = p[(s < 8 ? s : s + 8) * LDB2];
            }
#pragma unroll
            for (int mi = 0; mi < 4; ++mi)
#pragma unroll
                for (int ni = 0; ni < 4; ++ni)
                    acc[mi][ni] = __builtin_amdgcn_wmma_f32_16x16x32_bf16(
                        false, a[mi], false, b[ni],
                        (short)0, acc[mi][ni], false, false);
        }
        __syncthreads();   // all waves done reading buf before it is re-filled
    }

    // epilogue: D layout (VGPR r, lane l) -> (M=(l>=16?8:0)+r, N=l&15)
    const int rhalf = (lane >> 4) << 3;
    const int ocol  = lane & 15;
#pragma unroll
    for (int mi = 0; mi < 4; ++mi)
#pragma unroll
        for (int ni = 0; ni < 4; ++ni) {
            int col = bn0 + wcol * 64 + ni * 16 + ocol;
#pragma unroll
            for (int r = 0; r < 8; ++r) {
                int row = bm0 + wrow * 64 + mi * 16 + rhalf + r;
                out[(size_t)row * OUT_F + col] = (__bf16)acc[mi][ni][r];
            }
        }
}

// =====================================================================
// Fallback (fused, synchronous) if workspace too small for bf16 W'
// =====================================================================
#define BM 128
#define BN 128
#define BK 128
#define LDA (BK + 8)
#define LDB (BN + 8)

__global__ __launch_bounds__(256)
void wo_int4_gemm_fused(const __bf16* __restrict__ x,
                        const int*    __restrict__ qw,
                        const __bf16* __restrict__ sz,
                        __bf16*       __restrict__ out)
{
    __shared__ __bf16 Xs[BM * LDA];
    __shared__ __bf16 Ws[BK * LDB];

    const int t    = threadIdx.x;
    const int lane = t & 31;
    const int wid  = t >> 5;
    const int wrow = wid >> 2;
    const int wcol = wid & 3;
    const int bm0 = blockIdx.x * BM;
    const int bn0 = blockIdx.y * BN;

    v8f acc[4][2];
#pragma unroll
    for (int mi = 0; mi < 4; ++mi)
#pragma unroll
        for (int ni = 0; ni < 2; ++ni) {
            v8f z = {0.f, 0.f, 0.f, 0.f, 0.f, 0.f, 0.f, 0.f};
            acc[mi][ni] = z;
        }

    for (int kb = 0; kb < IN_F / BK; ++kb) {
        const int k0 = kb * BK;
#pragma unroll
        for (int i = 0; i < 8; ++i) {
            int c    = t + i * 256;
            int row  = c >> 4;
            int col8 = (c & 15) << 3;
            bf16x8 v = *(const bf16x8*)(x + (size_t)(bm0 + row) * IN_F + k0 + col8);
            *(bf16x8*)(Xs + row * LDA + col8) = v;
        }
        const __bf16* szg = sz + (size_t)kb * OUT_F * 2;
#pragma unroll 4
        for (int i = 0; i < 64; ++i) {
            int e = t + i * 256;
            int n = e >> 7;
            int k = e & 127;
            int q = qw[(size_t)(bn0 + n) * IN_F + k0 + k];
            uint32_t p = *(const uint32_t*)(szg + (size_t)(bn0 + n) * 2);
            union { uint32_t u; float f; } us, uz;
            us.u = p << 16;
            uz.u = p & 0xffff0000u;
            Ws[k * LDB + n] = (__bf16)((float)(q - 8) * us.f + uz.f);
        }
        __syncthreads();

#pragma unroll
        for (int kk = 0; kk < BK / 32; ++kk) {
            const int ahalf = (lane >> 4) << 3;
            v16bf a[4], b[2];
#pragma unroll
            for (int mi = 0; mi < 4; ++mi) {
                int row = wrow * 64 + mi * 16 + (lane & 15);
                const __bf16* p = Xs + row * LDA + kk * 32 + ahalf;
                bf16x8 lo = *(const bf16x8*)(p);
                bf16x8 hi = *(const bf16x8*)(p + 16);
                a[mi] = __builtin_shufflevector(lo, hi,
                        0,1,2,3,4,5,6,7,8,9,10,11,12,13,14,15);
            }
#pragma unroll
            for (int ni = 0; ni < 2; ++ni) {
                int krow = kk * 32 + ahalf;
                const __bf16* p = Ws + krow * LDB + wcol * 32 + ni * 16 + (lane & 15);
#pragma unroll
                for (int s = 0; s < 16; ++s)
                    b[ni][s] = p[(s < 8 ? s : s + 8) * LDB];
            }
#pragma unroll
            for (int mi = 0; mi < 4; ++mi)
#pragma unroll
                for (int ni = 0; ni < 2; ++ni)
                    acc[mi][ni] = __builtin_amdgcn_wmma_f32_16x16x32_bf16(
                        false, a[mi], false, b[ni],
                        (short)0, acc[mi][ni], false, false);
        }
        __syncthreads();
    }

    const int rhalf = (lane >> 4) << 3;
    const int ocol  = lane & 15;
#pragma unroll
    for (int mi = 0; mi < 4; ++mi)
#pragma unroll
        for (int ni = 0; ni < 2; ++ni) {
            int col = bn0 + wcol * 32 + ni * 16 + ocol;
#pragma unroll
            for (int r = 0; r < 8; ++r) {
                int row = bm0 + wrow * 64 + mi * 16 + rhalf + r;
                out[(size_t)row * OUT_F + col] = (__bf16)acc[mi][ni][r];
            }
        }
}

extern "C" void kernel_launch(void* const* d_in, const int* in_sizes, int n_in,
                              void* d_out, int out_size, void* d_ws, size_t ws_size,
                              hipStream_t stream) {
    const __bf16* x  = (const __bf16*)d_in[0];
    const int*    qw = (const int*)d_in[1];
    const __bf16* sz = (const __bf16*)d_in[2];
    __bf16*       o  = (__bf16*)d_out;

    const size_t WP_BYTES = (size_t)IN_F * OUT_F * sizeof(__bf16);  // 90.2 MB
    if (ws_size >= WP_BYTES) {
        __bf16* wP = (__bf16*)d_ws;
        dequant_transpose<<<dim3(OUT_F / 128, IN_F / 128), dim3(256), 0, stream>>>(qw, sz, wP);
        gemm_bf16_wmma_async<<<dim3(TOKENS / BM2, OUT_F / BN2), dim3(512), 0, stream>>>(x, wP, o);
    } else {
        wo_int4_gemm_fused<<<dim3(TOKENS / BM, OUT_F / BN), dim3(256), 0, stream>>>(x, qw, sz, o);
    }
}
